// Attention_51470888075483
// MI455X (gfx1250) — matrix operand order repaired and checked by reading, not compiled, not run on hardware
//
#include <hip/hip_runtime.h>
#include <hip/hip_bf16.h>
#include <math.h>

// ---------- problem constants (match reference) ----------
#define B_      128
#define N_      320
#define C_      256
#define H_      8
#define DK_     32
#define DV_     128
#define HW_     1536            // H_*(2*DK_+DV_)
#define DH_     1024            // H_*DV_
#define M_      (B_*N_)         // 40960 rows for both BN stages
#define BN_EPS_ 1e-5f
#define SCALE_  0.17677669529663687f   // 32^-0.5

typedef __attribute__((ext_vector_type(16))) _Float16 v16h;
typedef __attribute__((ext_vector_type(8)))  _Float16 v8h;
typedef __attribute__((ext_vector_type(8)))  float    v8f;
typedef __attribute__((ext_vector_type(4)))  unsigned int u32x4;
typedef __attribute__((ext_vector_type(8)))  int      i32x8;
typedef __attribute__((ext_vector_type(4)))  int      i32x4;

union F16x16 {
    v16h v;
    struct { v8h lo; v8h hi; } s;
    _Float16 e[16];
};

// ---------------------------------------------------------------------------
// CDNA5 async global->LDS copy (16B per lane, ASYNCcnt-tracked)
// ---------------------------------------------------------------------------
__device__ __forceinline__ void async_copy_b128(unsigned lds_off,
                                                unsigned long long gaddr) {
    asm volatile("global_load_async_to_lds_b128 %0, %1, off"
                 :: "v"(lds_off), "v"(gaddr) : "memory");
}
__device__ __forceinline__ void wait_async0() {
#if __has_builtin(__builtin_amdgcn_s_wait_asynccnt)
    __builtin_amdgcn_s_wait_asynccnt(0);
#else
    asm volatile("s_wait_asynccnt 0x0" ::: "memory");
#endif
}
__device__ __forceinline__ void wait_tensor0() {
#if __has_builtin(__builtin_amdgcn_s_wait_tensorcnt)
    __builtin_amdgcn_s_wait_tensorcnt(0);
#else
    asm volatile("s_wait_tensorcnt 0x0" ::: "memory");
#endif
}

// ---------------------------------------------------------------------------
// TDM: 1-D contiguous tile load via Tensor Data Mover (TENSORcnt-tracked).
// D# per CDNA5 ISA ch.8: group0 = {count=1, lds_addr, global_addr[56:0], type=2},
// group1 = {data_size=8B, tensor_dim0 = nElems8B, tile_dim0 = nElems8B,
//           tensor_dim1 = tile_dim1 = 1, stride0 = nElems8B}.
// Issue from ONE wave only (EXEC is ignored; every wave would re-issue the DMA).
// ---------------------------------------------------------------------------
#if __has_builtin(__builtin_amdgcn_tensor_load_to_lds)
#define HAVE_TDM_ 1
__device__ __forceinline__ void tdm_load_1d(unsigned lds_off,
                                            unsigned long long gaddr,
                                            unsigned nElems8B) {
    u32x4 g0 = {};
    g0[0] = 1u;                                               // count=1 (valid user D#)
    g0[1] = lds_off;                                          // lds_addr (bytes)
    g0[2] = (unsigned)(gaddr & 0xffffffffu);                  // global_addr[31:0]
    g0[3] = (unsigned)((gaddr >> 32) & 0x01ffffffu) | (2u << 30); // addr[56:32] | type=2
    i32x8 g1 = {};
    g1[0] = (int)(3u << 16);                                  // data_size = 8 bytes
    g1[1] = (int)((nElems8B & 0xffffu) << 16);                // tensor_dim0[15:0]
    g1[2] = (int)(((nElems8B >> 16) & 0xffffu) | (1u << 16)); // tensor_dim0[31:16], tensor_dim1=1
    g1[3] = (int)((nElems8B & 0xffffu) << 16);                // tile_dim0
    g1[4] = 1;                                                // tile_dim1 = 1, tile_dim2 = 0
    g1[5] = (int)nElems8B;                                    // tensor_dim0_stride[31:0]
    i32x4 g2 = {}, g3 = {};
#if __clang_major__ >= 23
    i32x8 g4 = {};
    __builtin_amdgcn_tensor_load_to_lds(g0, g1, g2, g3, g4, 0);
#else
    __builtin_amdgcn_tensor_load_to_lds(g0, g1, g2, g3, 0);
#endif
}
#else
#define HAVE_TDM_ 0
#endif

// ---------------------------------------------------------------------------
// f32 -> f16 conversion (x, Wqkv, Wproj)
// ---------------------------------------------------------------------------
__global__ void cvt_f32_to_f16(const float* __restrict__ src,
                               _Float16* __restrict__ dst, long long n) {
    long long i = (long long)blockIdx.x * blockDim.x + threadIdx.x;
    if (i < n) dst[i] = (_Float16)src[i];
}

// ---------------------------------------------------------------------------
// Generic WMMA GEMM: C[M,N] = A[M,K] * B[N,K]^T   (A,B f16 K-contiguous, C f32)
// block = 128 threads (4 waves as 2x2), tile = 64x64, K step 32.
// ---------------------------------------------------------------------------
__global__ void gemm_wmma_f16(const _Float16* __restrict__ A,
                              const _Float16* __restrict__ Bm,
                              float* __restrict__ C,
                              int M, int N, int K) {
    const int lane = threadIdx.x & 31;
    const int w    = threadIdx.x >> 5;
    const int wr   = w >> 1, wc = w & 1;
    const int m0   = blockIdx.x * 64 + wr * 32;
    const int n0   = blockIdx.y * 64 + wc * 32;
    const int lx   = lane & 15, lh = lane >> 4;

    v8f acc[2][2] = {};
    for (int k0 = 0; k0 < K; k0 += 32) {
        F16x16 a[2], b[2];
#pragma unroll
        for (int i = 0; i < 2; ++i) {
            const _Float16* ap = A + (long long)(m0 + i * 16 + lx) * K + k0;
            a[i].s.lo = *(const v8h*)(ap + lh * 16);
            a[i].s.hi = *(const v8h*)(ap + lh * 16 + 8);
            // unconditional speculative prefetch of next K-step (translation
            // failures are silently dropped on CDNA5) -> global_prefetch_b8
            __builtin_prefetch(ap + 32, 0, 3);
            const _Float16* bp = Bm + (long long)(n0 + i * 16 + lx) * K + k0 + lh * 16;
            b[i].v = *(const v16h*)bp;
        }
#pragma unroll
        for (int i = 0; i < 2; ++i)
#pragma unroll
            for (int j = 0; j < 2; ++j)
                acc[i][j] = __builtin_amdgcn_wmma_f32_16x16x32_f16(
                    false, a[i].v, false, b[j].v, (short)0, acc[i][j], false, false);
    }
#pragma unroll
    for (int i = 0; i < 2; ++i)
#pragma unroll
        for (int j = 0; j < 2; ++j)
#pragma unroll
            for (int r = 0; r < 8; ++r) {
                int row = m0 + i * 16 + r + lh * 8;
                int col = n0 + j * 16 + lx;
                C[(long long)row * N + col] = acc[i][j][r];
            }
}

// ---------------------------------------------------------------------------
// Deterministic column statistics, stage 1: per-(row-chunk) partial sums.
// ---------------------------------------------------------------------------
__global__ void col_stats_partial(const float* __restrict__ X,
                                  float* __restrict__ psum,
                                  float* __restrict__ psumsq,
                                  int Ncols, int rowsPerBlock) {
    __shared__ float red[512];
    const int t  = threadIdx.x;
    const int c  = blockIdx.x * 64 + (t & 63);
    const int rg = t >> 6;
    const long long r0 = (long long)blockIdx.y * rowsPerBlock;
    float s = 0.f, ss = 0.f;
    for (int r = rg; r < rowsPerBlock; r += 4) {
        float v = X[(r0 + r) * Ncols + c];
        s += v; ss += v * v;
    }
    red[t] = s; red[256 + t] = ss;
    __syncthreads();
    if (t < 64) {
        float a  = red[t] + red[t + 64] + red[t + 128] + red[t + 192];
        float b  = red[256 + t] + red[256 + t + 64] + red[256 + t + 128] + red[256 + t + 192];
        psum[(long long)blockIdx.y * Ncols + c]   = a;
        psumsq[(long long)blockIdx.y * Ncols + c] = b;
    }
}

__global__ void col_stats_reduce(const float* __restrict__ psum,
                                 const float* __restrict__ psumsq,
                                 float* __restrict__ sums,
                                 float* __restrict__ sumsq,
                                 int Ncols, int chunks) {
    int c = blockIdx.x * 256 + threadIdx.x;
    if (c >= Ncols) return;
    float s = 0.f, ss = 0.f;
    for (int i = 0; i < chunks; ++i) {
        s  += psum[(long long)i * Ncols + c];
        ss += psumsq[(long long)i * Ncols + c];
    }
    sums[c] = s; sumsq[c] = ss;
}

// ---------------------------------------------------------------------------
// BN(train) on qkv + scatter into WMMA-friendly f16 q/k/v buffers.
//   qh, kh : [b*H+h][n][dk]  (dk contiguous; per-(b,h) block is contiguous)
//   vt     : [b*H+h][dv][n]  (n contiguous;  per-(b,h) block is contiguous)
// ---------------------------------------------------------------------------
__global__ void bn_scatter_qkv(const float* __restrict__ qkv,
                               const float* __restrict__ sums,
                               const float* __restrict__ sumsq,
                               const float* __restrict__ g1,
                               const float* __restrict__ b1,
                               _Float16* __restrict__ qh,
                               _Float16* __restrict__ kh,
                               _Float16* __restrict__ vt) {
    long long idx = (long long)blockIdx.x * blockDim.x + threadIdx.x;
    const long long total = (long long)M_ * HW_;
    if (idx >= total) return;
    int j = (int)(idx % HW_);
    long long m = idx / HW_;
    const float invM = 1.0f / (float)M_;
    float mean = sums[j] * invM;
    float var  = sumsq[j] * invM - mean * mean;
    float val  = (qkv[idx] - mean) * rsqrtf(var + BN_EPS_) * g1[j] + b1[j];
    int b = (int)(m / N_), n = (int)(m % N_);
    int hh = j / (2 * DK_ + DV_), rem = j % (2 * DK_ + DV_);
    long long bh = (long long)b * H_ + hh;
    if (rem < DK_)
        qh[(bh * N_ + n) * DK_ + rem] = (_Float16)val;
    else if (rem < 2 * DK_)
        kh[(bh * N_ + n) * DK_ + (rem - DK_)] = (_Float16)val;
    else
        vt[(bh * DV_ + (rem - 2 * DK_)) * N_ + n] = (_Float16)val;
}

// ---------------------------------------------------------------------------
// Attention: one workgroup per (64-row n-block, head, batch). 128 threads.
// LDS carve (dynamic, 225,280 B of the 320 KB WGP LDS):
//   [0      , 20480 )  K tile  f16 [320][32]   <- async global->LDS (ASYNCcnt)
//   [20480  , 102400)  V tile  f16 [128][320]  <- TDM descriptor (TENSORcnt)
//   [102400 , 184320)  scores  f32 [64][320]
//   [184320 , 225280)  probs   f16 [64][320]
// ---------------------------------------------------------------------------
#define K_OFF_  0
#define V_OFF_  20480
#define S_OFF_  102400
#define P_OFF_  184320
#define LDS_SZ_ 225280

__global__ void attn_wmma(const _Float16* __restrict__ qh,
                          const _Float16* __restrict__ kh,
                          const _Float16* __restrict__ vt,
                          const float* __restrict__ ab,
                          const int* __restrict__ idxs,
                          _Float16* __restrict__ geluh,
                          int n_off) {
    extern __shared__ char smem[];
    _Float16* kT = (_Float16*)(smem + K_OFF_);
    _Float16* vT = (_Float16*)(smem + V_OFF_);
    float*    sc = (float*)   (smem + S_OFF_);
    _Float16* pT = (_Float16*)(smem + P_OFF_);

    const int b  = blockIdx.z, h = blockIdx.y, n0 = blockIdx.x * 64;
    const int t  = threadIdx.x;
    const int lane = t & 31, w = t >> 5;
    const int lx = lane & 15, lh = lane >> 4;
    const long long bh = (long long)b * H_ + h;

    const unsigned kBase = (unsigned)(uintptr_t)kT;
    const unsigned vBase = (unsigned)(uintptr_t)vT;
    unsigned long long kSrc = (unsigned long long)(kh + bh * (N_ * DK_));
    unsigned long long vSrc = (unsigned long long)(vt + bh * (DV_ * N_));

    // ---- V tile: one TDM descriptor issued by wave 0 (81920 B contiguous) ----
#if HAVE_TDM_
    if (w == 0) {
        if (lane == 0) tdm_load_1d(vBase, vSrc, (DV_ * N_ * 2) / 8);
        wait_tensor0();
    }
#else
    for (int i = t; i < (DV_ * N_ * 2) / 16; i += 128)
        async_copy_b128(vBase + i * 16, vSrc + i * 16);
#endif

    // ---- K tile: per-lane async global->LDS (20480 B contiguous) ----
    for (int i = t; i < (N_ * DK_ * 2) / 16; i += 128)        // 1280 chunks
        async_copy_b128(kBase + i * 16, kSrc + i * 16);

    // Q fragment (16 rows x 32 K) for this wave, from global (overlaps copies).
    F16x16 qf;
    {
        const _Float16* qp = qh + (bh * N_ + (n0 + w * 16 + lx)) * DK_;
        qf.s.lo = *(const v8h*)(qp + lh * 16);
        qf.s.hi = *(const v8h*)(qp + lh * 16 + 8);
    }

    wait_async0();
    __syncthreads();   // makes TDM-filled V + async-filled K visible to all waves

    // ---- scores: S = QK^T * scale + gathered bias -> LDS (f32) ----
    for (int m0 = 0; m0 < N_; m0 += 16) {
        F16x16 kf;
        kf.v = *(const v16h*)(kT + (m0 + lx) * DK_ + lh * 16);
        v8f z = {};
        v8f s = __builtin_amdgcn_wmma_f32_16x16x32_f16(
            false, qf.v, false, kf.v, (short)0, z, false, false);
#pragma unroll
        for (int r = 0; r < 8; ++r) {
            int nl = w * 16 + r + lh * 8;
            int n  = n0 + nl;
            int m  = m0 + lx;
            float bias = ab[h * n_off + idxs[n * N_ + m]];
            sc[nl * N_ + m] = s[r] * SCALE_ + bias;
        }
    }
    __syncthreads();

    // ---- row softmax (f32), emit normalized probs as f16 tile ----
    if (t < 64) {
        float* row = sc + t * N_;
        _Float16* prow = pT + t * N_;
        float mx = -3.402823466e38f;
        for (int m = 0; m < N_; ++m) mx = fmaxf(mx, row[m]);
        float sum = 0.f;
        for (int m = 0; m < N_; ++m) { float e = expf(row[m] - mx); row[m] = e; sum += e; }
        float inv = 1.0f / sum;
        for (int m = 0; m < N_; ++m) prow[m] = (_Float16)(row[m] * inv);
    }
    __syncthreads();

    // ---- O = P V from LDS tiles, fused exact GeLU ----
    v8f acc[8] = {};
    for (int ks = 0; ks < N_; ks += 32) {
        F16x16 pf;
        {
            const _Float16* prow = pT + (w * 16 + lx) * N_ + ks;
            pf.s.lo = *(const v8h*)(prow + lh * 16);
            pf.s.hi = *(const v8h*)(prow + lh * 16 + 8);
        }
#pragma unroll
        for (int db = 0; db < 8; ++db) {
            F16x16 vf;
            vf.v = *(const v16h*)(vT + (db * 16 + lx) * N_ + ks + lh * 16);
            acc[db] = __builtin_amdgcn_wmma_f32_16x16x32_f16(
                false, pf.v, false, vf.v, (short)0, acc[db], false, false);
        }
    }
#pragma unroll
    for (int db = 0; db < 8; ++db)
#pragma unroll
        for (int r = 0; r < 8; ++r) {
            int n = n0 + w * 16 + r + lh * 8;
            int d = db * 16 + lx;
            float x = acc[db][r];
            float g = 0.5f * x * (1.0f + erff(x * 0.70710678118654752f));
            geluh[((long long)b * N_ + n) * DH_ + h * DV_ + d] = (_Float16)g;
        }
}

// ---------------------------------------------------------------------------
// Final BN(train) -> f32 output
// ---------------------------------------------------------------------------
__global__ void bn_final(const float* __restrict__ X,
                         const float* __restrict__ sums,
                         const float* __restrict__ sumsq,
                         const float* __restrict__ g2,
                         const float* __restrict__ b2,
                         float* __restrict__ out) {
    long long idx = (long long)blockIdx.x * blockDim.x + threadIdx.x;
    const long long total = (long long)M_ * C_;
    if (idx >= total) return;
    int c = (int)(idx & (C_ - 1));
    const float invM = 1.0f / (float)M_;
    float mean = sums[c] * invM;
    float var  = sumsq[c] * invM - mean * mean;
    out[idx] = (X[idx] - mean) * rsqrtf(var + BN_EPS_) * g2[c] + b2[c];
}

// ---------------------------------------------------------------------------
// Host launcher
// ---------------------------------------------------------------------------
extern "C" void kernel_launch(void* const* d_in, const int* in_sizes, int n_in,
                              void* d_out, int out_size, void* d_ws, size_t ws_size,
                              hipStream_t stream) {
    const float* x     = (const float*)d_in[0];
    const float* Wqkv  = (const float*)d_in[1];
    const float* g1    = (const float*)d_in[2];
    const float* b1    = (const float*)d_in[3];
    const float* ab    = (const float*)d_in[4];
    const float* Wproj = (const float*)d_in[5];
    const float* g2    = (const float*)d_in[6];
    const float* b2    = (const float*)d_in[7];
    const int*   idxs  = (const int*)d_in[8];
    const int n_off = in_sizes[4] / H_;

    char* base = (char*)d_ws;
    size_t off = 0;
    auto take = [&](size_t bytes) -> void* {
        void* p = base + off;
        off += (bytes + 255) & ~(size_t)255;
        return p;
    };
    _Float16* xh     = (_Float16*)take((size_t)M_ * C_ * 2);
    _Float16* wqkvh  = (_Float16*)take((size_t)HW_ * C_ * 2);
    _Float16* wprojh = (_Float16*)take((size_t)C_ * DH_ * 2);
    float*    qkv    = (float*)   take((size_t)M_ * HW_ * 4);
    _Float16* qh     = (_Float16*)take((size_t)B_ * H_ * N_ * DK_ * 2);
    _Float16* kh     = (_Float16*)take((size_t)B_ * H_ * N_ * DK_ * 2);
    _Float16* vt     = (_Float16*)take((size_t)B_ * H_ * DV_ * N_ * 2);
    _Float16* geluh  = (_Float16*)take((size_t)M_ * DH_ * 2);
    float*    psum   = (float*)   take((size_t)160 * HW_ * 4);
    float*    psumsq = (float*)   take((size_t)160 * HW_ * 4);
    float*    sums1  = (float*)take(HW_ * 4);
    float*    ssq1   = (float*)take(HW_ * 4);
    float*    sums2  = (float*)take(C_ * 4);
    float*    ssq2   = (float*)take(C_ * 4);
    float*    out2   = qkv;   // reuse: qkv dead after bn_scatter_qkv

    // 1) f32 -> f16 operand conversion
    {
        long long n;
        n = (long long)M_ * C_;
        cvt_f32_to_f16<<<(unsigned)((n + 255) / 256), 256, 0, stream>>>(x, xh, n);
        n = (long long)HW_ * C_;
        cvt_f32_to_f16<<<(unsigned)((n + 255) / 256), 256, 0, stream>>>(Wqkv, wqkvh, n);
        n = (long long)C_ * DH_;
        cvt_f32_to_f16<<<(unsigned)((n + 255) / 256), 256, 0, stream>>>(Wproj, wprojh, n);
    }

    // 2) QKV GEMM
    gemm_wmma_f16<<<dim3(M_ / 64, HW_ / 64), 128, 0, stream>>>(xh, wqkvh, qkv, M_, HW_, C_);

    // 3) BN1 statistics
    col_stats_partial<<<dim3(HW_ / 64, 160), 256, 0, stream>>>(qkv, psum, psumsq, HW_, M_ / 160);
    col_stats_reduce<<<(HW_ + 255) / 256, 256, 0, stream>>>(psum, psumsq, sums1, ssq1, HW_, 160);

    // 4) BN1 apply + scatter
    {
        long long total = (long long)M_ * HW_;
        bn_scatter_qkv<<<(unsigned)((total + 255) / 256), 256, 0, stream>>>(
            qkv, sums1, ssq1, g1, b1, qh, kh, vt);
    }

    // 5) attention (TDM V-tile + async K-tile staging into 220 KB LDS)
    attn_wmma<<<dim3(N_ / 64, H_, B_), 128, LDS_SZ_, stream>>>(
        qh, kh, vt, ab, idxs, geluh, n_off);

    // 6) projection GEMM
    gemm_wmma_f16<<<dim3(M_ / 64, C_ / 64), 128, 0, stream>>>(geluh, wprojh, out2, M_, C_, DH_);

    // 7) BN2 statistics + apply -> d_out
    col_stats_partial<<<dim3(C_ / 64, 160), 256, 0, stream>>>(out2, psum, psumsq, C_, M_ / 160);
    col_stats_reduce<<<1, 256, 0, stream>>>(psum, psumsq, sums2, ssq2, C_, 160);
    {
        long long total = (long long)M_ * C_;
        bn_final<<<(unsigned)((total + 255) / 256), 256, 0, stream>>>(
            out2, sums2, ssq2, g2, b2, (float*)d_out);
    }
}
